// TransformerBlock_64458869178977
// MI455X (gfx1250) — matrix operand order repaired and checked by reading, not compiled, not run on hardware
//
#include <hip/hip_runtime.h>

// ---------------- problem constants ----------------
constexpr int BB  = 2;
constexpr int TT  = 1024;
constexpr int DD  = 1024;
constexpr int HH  = 16;
constexpr int HDD = 64;          // head dim
constexpr int EE  = 8;           // experts
constexpr int FFF = 4096;        // FF dim
constexpr int NN  = BB * TT;     // 2048 tokens
constexpr int CAP = NN;          // worst-case tokens per expert
constexpr float EPSF = 1e-5f;

// ---------------- WMMA types ----------------
typedef __attribute__((ext_vector_type(16))) __bf16 v16bf;
typedef __attribute__((ext_vector_type(8)))  float  v8f;

union BF16x16 {
  unsigned short u[16];
  unsigned int   d[8];
  v16bf          v;
};

__device__ __forceinline__ unsigned short f2bf(float f) {
  unsigned int x = __float_as_uint(f);
  unsigned int r = x + 0x7fffu + ((x >> 16) & 1u);   // round-to-nearest-even
  return (unsigned short)(r >> 16);
}
__device__ __forceinline__ float bf2f(unsigned short h) {
  return __uint_as_float(((unsigned int)h) << 16);
}

#define WMMA_BF16(A, B, C) \
  __builtin_amdgcn_wmma_f32_16x16x32_bf16(false, (A), false, (B), (short)0, (C), false, false)

// ---------------- small utility kernels ----------------
__global__ void zero_ints(int* p, int n) {
  int i = blockIdx.x * blockDim.x + threadIdx.x;
  if (i < n) p[i] = 0;
}

// fp32 [K,N] -> bf16 transposed [N,K]; 32x32 LDS tile; blockIdx.z selects expert slab
__global__ __launch_bounds__(256)
void transpose_cvt(const float* __restrict__ W, unsigned short* __restrict__ Wt,
                   int K, int N) {
  __shared__ unsigned short tile[32][33];
  const size_t slab = (size_t)blockIdx.z * K * N;
  const float* Ws = W + slab;
  unsigned short* Wts = Wt + slab;
  const int n0 = blockIdx.x * 32, k0 = blockIdx.y * 32;
  const int tx = threadIdx.x & 31, ty = threadIdx.x >> 5;   // 32 x 8
#pragma unroll
  for (int i = 0; i < 32; i += 8)
    tile[ty + i][tx] = f2bf(Ws[(size_t)(k0 + ty + i) * N + n0 + tx]);
  __syncthreads();
#pragma unroll
  for (int i = 0; i < 32; i += 8)
    Wts[(size_t)(n0 + ty + i) * K + k0 + tx] = tile[tx][ty + i];
}

// one block per row; x row (f32) -> bf16 normalized row
__global__ __launch_bounds__(256)
void rmsnorm_k(const float* __restrict__ x, const float* __restrict__ scale,
               unsigned short* __restrict__ out) {
  __shared__ float red[256];
  const int row = blockIdx.x;
  const int tid = threadIdx.x;
  const float* xr = x + (size_t)row * DD;
  float s = 0.f;
  for (int d = tid; d < DD; d += 256) { float v = xr[d]; s += v * v; }
  red[tid] = s;
  __syncthreads();
  for (int st = 128; st; st >>= 1) {
    if (tid < st) red[tid] += red[tid + st];
    __syncthreads();
  }
  const float rms = sqrtf(red[0] / (float)DD);
  const float inv = 1.f / (rms + EPSF);
  for (int d = tid; d < DD; d += 256)
    out[(size_t)row * DD + d] = f2bf(xr[d] * inv * scale[d]);
}

// ---------------- GEMM: C[M,N] = A(bf16)[M,K] * Bt(bf16)[N,K]^T (+bias, +resid) ----------------
// block = 128 = 4 waves; wave tile = 16(m) x 64(n); grid = (N/64, M/64)
__global__ __launch_bounds__(128)
void gemm_bf16(const unsigned short* __restrict__ A, const unsigned short* __restrict__ Bt,
               const float* __restrict__ bias, const float* __restrict__ resid,
               float* __restrict__ Cf, int M, int N, int K) {
  const int lane = threadIdx.x & 31;
  const int wave = threadIdx.x >> 5;
  const int mBase = blockIdx.y * 64 + wave * 16;
  const int nBase = blockIdx.x * 64;
  if (mBase >= M) return;

  const int m   = lane & 15;
  const int hf  = lane >> 4;
  const int kbA = hf * 16;
  const int kbB = hf * 16;
  const int n   = lane & 15;

  v8f acc[4] = {{}, {}, {}, {}};
  const unsigned short* Arow = A + (size_t)(mBase + m) * K;
  const unsigned short* Bt0  = Bt + (size_t)(nBase + n) * K;

  for (int k0 = 0; k0 < K; k0 += 32) {
    __builtin_prefetch(Arow + k0 + 512, 0, 1);
    __builtin_prefetch(Bt0 + k0 + 512, 0, 1);
    BF16x16 af, bf0, bf1, bf2, bf3;
#pragma unroll
    for (int j = 0; j < 4; ++j) {
      const int o = k0 + kbA + 2 * j;
      af.d[j]     = *(const unsigned int*)(Arow + o);
      af.d[4 + j] = *(const unsigned int*)(Arow + 8 + o);
    }
#pragma unroll
    for (int j = 0; j < 8; ++j) {
      const int o = k0 + kbB + 2 * j;
      bf0.d[j] = *(const unsigned int*)(Bt0 + o);
      bf1.d[j] = *(const unsigned int*)(Bt0 + (size_t)16 * K + o);
      bf2.d[j] = *(const unsigned int*)(Bt0 + (size_t)32 * K + o);
      bf3.d[j] = *(const unsigned int*)(Bt0 + (size_t)48 * K + o);
    }
    acc[0] = WMMA_BF16(af.v, bf0.v, acc[0]);
    acc[1] = WMMA_BF16(af.v, bf1.v, acc[1]);
    acc[2] = WMMA_BF16(af.v, bf2.v, acc[2]);
    acc[3] = WMMA_BF16(af.v, bf3.v, acc[3]);
  }

#pragma unroll
  for (int t = 0; t < 4; ++t) {
    const int col = nBase + t * 16 + n;
    const float bv = bias ? bias[col] : 0.f;
#pragma unroll
    for (int r = 0; r < 8; ++r) {
      const int row = mBase + r + 8 * hf;
      const size_t idx = (size_t)row * N + col;
      float v = acc[t][r] + bv;
      if (resid) v += resid[idx];
      Cf[idx] = v;
    }
  }
}

// ---------------- RoPE + pack q/k/v to bf16 ----------------
__global__ __launch_bounds__(64)
void rope_pack(const float* __restrict__ qkv, unsigned short* __restrict__ qb,
               unsigned short* __restrict__ kb, unsigned short* __restrict__ vtb) {
  const int t = blockIdx.x, h = blockIdx.y, b = blockIdx.z, hd = threadIdx.x;
  const float* base = qkv + (size_t)(b * TT + t) * (3 * DD) + h * HDD;
  const float qv = base[hd];
  const float kv = base[DD + hd];
  const float vv = base[2 * DD + hd];
  const int  hlf = hd & 31;
  const bool hi  = hd >= 32;
  const float oq = base[hi ? hd - 32 : hd + 32];
  const float ok = base[DD + (hi ? hd - 32 : hd + 32)];
  const float theta = __powf(10000.f, -2.f * (float)hlf / (float)HDD);
  const float ang = (float)t * theta;
  const float s = __sinf(ang), c = __cosf(ang);
  const float qr = hi ? (oq * s + qv * c) : (qv * c - oq * s);
  const float kr = hi ? (ok * s + kv * c) : (kv * c - ok * s);
  const size_t o = ((size_t)(b * HH + h) * TT + t) * HDD + hd;
  qb[o] = f2bf(qr * 0.125f);       // fold 1/sqrt(HD) into q
  kb[o] = f2bf(kr);
  vtb[((size_t)(b * HH + h) * HDD + hd) * TT + t] = f2bf(vv);  // V transposed
}

// ---------------- flash attention: one wave per (b,h,16-query tile) ----------------
__global__ __launch_bounds__(32)
void flash_attn(const unsigned short* __restrict__ qb, const unsigned short* __restrict__ kb,
                const unsigned short* __restrict__ vtb, unsigned short* __restrict__ ob) {
  __shared__ alignas(16) unsigned short Plds[16 * 32];
  const int lane = threadIdx.x;
  const int q0 = blockIdx.x * 16, h = blockIdx.y, b = blockIdx.z;
  const int m = lane & 15;
  const int hf = lane >> 4;
  const int kbA = hf * 16;
  const int kbB = hf * 16;
  const size_t bh = (size_t)(b * HH + h);

  BF16x16 qa0, qa1;
  {
    const unsigned short* Qrow = qb + (bh * TT + q0 + m) * HDD;
#pragma unroll
    for (int j = 0; j < 4; ++j) {
      const int o = kbA + 2 * j;
      qa0.d[j]     = *(const unsigned int*)(Qrow + o);
      qa0.d[4 + j] = *(const unsigned int*)(Qrow + 8 + o);
      qa1.d[j]     = *(const unsigned int*)(Qrow + 32 + o);
      qa1.d[4 + j] = *(const unsigned int*)(Qrow + 40 + o);
    }
  }

  v8f o0 = {}, o1 = {}, o2 = {}, o3 = {};
  float mrun[8], lrun[8];
#pragma unroll
  for (int r = 0; r < 8; ++r) { mrun[r] = -1e30f; lrun[r] = 0.f; }

  const int tEnd = q0 + 16;
  for (int t0 = 0; t0 < tEnd; t0 += 32) {
    v8f s0 = {}, s1 = {};
    {
      const unsigned short* Krow = kb + (bh * TT + t0 + (lane & 15)) * HDD;
      const unsigned short* Krow2 = Krow + (size_t)16 * HDD;
      BF16x16 k0f, k1f, k2f, k3f;
#pragma unroll
      for (int j = 0; j < 8; ++j) {
        const int o = kbB + 2 * j;
        k0f.d[j] = *(const unsigned int*)(Krow + o);
        k1f.d[j] = *(const unsigned int*)(Krow + 32 + o);
        k2f.d[j] = *(const unsigned int*)(Krow2 + o);
        k3f.d[j] = *(const unsigned int*)(Krow2 + 32 + o);
      }
      s0 = WMMA_BF16(qa0.v, k0f.v, s0);
      s0 = WMMA_BF16(qa1.v, k1f.v, s0);
      s1 = WMMA_BF16(qa0.v, k2f.v, s1);
      s1 = WMMA_BF16(qa1.v, k3f.v, s1);
    }

#pragma unroll
    for (int r = 0; r < 8; ++r) {
      const int tq  = q0 + r + 8 * hf;
      const int tk0 = t0 + (lane & 15);
      const int tk1 = t0 + 16 + (lane & 15);
      float e0 = (tk0 <= tq) ? s0[r] : -1e30f;
      float e1 = (tk1 <= tq) ? s1[r] : -1e30f;
      float tm = fmaxf(e0, e1);
#pragma unroll
      for (int off = 8; off; off >>= 1) tm = fmaxf(tm, __shfl_xor(tm, off, 32));
      const float mn = fmaxf(mrun[r], tm);
      const float alpha = __expf(mrun[r] - mn);
      const float p0 = __expf(e0 - mn);
      const float p1 = __expf(e1 - mn);
      float ls = p0 + p1;
#pragma unroll
      for (int off = 8; off; off >>= 1) ls += __shfl_xor(ls, off, 32);
      lrun[r] = lrun[r] * alpha + ls;
      mrun[r] = mn;
      o0[r] *= alpha; o1[r] *= alpha; o2[r] *= alpha; o3[r] *= alpha;
      const int row = r + 8 * hf;
      Plds[row * 32 + (lane & 15)]      = f2bf(p0);
      Plds[row * 32 + 16 + (lane & 15)] = f2bf(p1);
    }
    __syncthreads();

    BF16x16 pa;
#pragma unroll
    for (int j = 0; j < 4; ++j) {
      const int o = kbA + 2 * j;
      pa.d[j]     = *(const unsigned int*)(Plds + m * 32 + o);
      pa.d[4 + j] = *(const unsigned int*)(Plds + m * 32 + 8 + o);
    }
#pragma unroll
    for (int ti = 0; ti < 4; ++ti) {
      const unsigned short* Vrow = vtb + (bh * HDD + 16 * ti + (lane & 15)) * TT;
      BF16x16 vf;
#pragma unroll
      for (int j = 0; j < 8; ++j)
        vf.d[j] = *(const unsigned int*)(Vrow + t0 + kbB + 2 * j);
      if (ti == 0) o0 = WMMA_BF16(pa.v, vf.v, o0);
      else if (ti == 1) o1 = WMMA_BF16(pa.v, vf.v, o1);
      else if (ti == 2) o2 = WMMA_BF16(pa.v, vf.v, o2);
      else o3 = WMMA_BF16(pa.v, vf.v, o3);
    }
    __syncthreads();
  }

#pragma unroll
  for (int r = 0; r < 8; ++r) {
    const int tq = q0 + r + 8 * hf;
    const float inv = 1.f / lrun[r];
    const size_t rowo = (size_t)(b * TT + tq) * DD + h * HDD + (lane & 15);
    ob[rowo]      = f2bf(o0[r] * inv);
    ob[rowo + 16] = f2bf(o1[r] * inv);
    ob[rowo + 32] = f2bf(o2[r] * inv);
    ob[rowo + 48] = f2bf(o3[r] * inv);
  }
}

// ---------------- MoE gate: top-2, softmax, expert lists ----------------
__global__ __launch_bounds__(32)
void gate_topk(const unsigned short* __restrict__ Xbf, const float* __restrict__ wg,
               const float* __restrict__ bg, int* __restrict__ cnt,
               int* __restrict__ lists, float* __restrict__ probs) {
  const int tok = blockIdx.x;
  const int lane = threadIdx.x;
  float acc[EE];
#pragma unroll
  for (int e = 0; e < EE; ++e) acc[e] = 0.f;
  const unsigned short* xr = Xbf + (size_t)tok * DD;
  for (int d = lane; d < DD; d += 32) {
    const float xv = bf2f(xr[d]);
    const float* wr = wg + (size_t)d * EE;
#pragma unroll
    for (int e = 0; e < EE; ++e) acc[e] += xv * wr[e];
  }
#pragma unroll
  for (int e = 0; e < EE; ++e)
#pragma unroll
    for (int off = 16; off; off >>= 1) acc[e] += __shfl_xor(acc[e], off, 32);

  if (lane == 0) {
    float lg[EE];
#pragma unroll
    for (int e = 0; e < EE; ++e) lg[e] = acc[e] + bg[e];
    int i0 = 0;
    for (int e = 1; e < EE; ++e) if (lg[e] > lg[i0]) i0 = e;
    int i1 = -1;
    for (int e = 0; e < EE; ++e) {
      if (e == i0) continue;
      if (i1 < 0 || lg[e] > lg[i1]) i1 = e;
    }
    const float p0 = 1.f / (1.f + __expf(lg[i1] - lg[i0]));
    const float p1 = 1.f - p0;
    const int pos0 = atomicAdd(&cnt[i0], 1);
    lists[i0 * CAP + pos0] = tok; probs[i0 * CAP + pos0] = p0;
    const int pos1 = atomicAdd(&cnt[i1], 1);
    lists[i1 * CAP + pos1] = tok; probs[i1 * CAP + pos1] = p1;
  }
}

// ---------------- expert stage 1: h = silu(x*w1+b1)*(x*w2+b2); wave tile 16x32 per weight ----------------
// grid = (FFF/128, CAP/16, EE); block 128
__global__ __launch_bounds__(128)
void moe_ff1(const unsigned short* __restrict__ Xbf,
             const unsigned short* __restrict__ w1t, const float* __restrict__ b1,
             const unsigned short* __restrict__ w2t, const float* __restrict__ b2,
             const int* __restrict__ lists, const int* __restrict__ cnt,
             unsigned short* __restrict__ Hbf) {
  const int e = blockIdx.z;
  const int count = cnt[e];
  const int mBase = blockIdx.y * 16;
  if (mBase >= count) return;
  const int lane = threadIdx.x & 31;
  const int wave = threadIdx.x >> 5;
  const int nBase = (blockIdx.x * 4 + wave) * 32;
  const int m = lane & 15, hf = lane >> 4, kbA = hf * 16, kbB = hf * 16, n = lane & 15;

  const int row = mBase + m;
  const int tok = lists[e * CAP + (row < count ? row : 0)];
  const unsigned short* Arow = Xbf + (size_t)tok * DD;
  const unsigned short* B1r = w1t + ((size_t)e * FFF + nBase + n) * DD;
  const unsigned short* B2r = w2t + ((size_t)e * FFF + nBase + n) * DD;

  v8f a1a = {}, a1b = {}, a2a = {}, a2b = {};
  for (int k0 = 0; k0 < DD; k0 += 32) {
    __builtin_prefetch(B1r + k0 + 512, 0, 1);
    __builtin_prefetch(B2r + k0 + 512, 0, 1);
    BF16x16 af, b1f0, b1f1, b2f0, b2f1;
#pragma unroll
    for (int j = 0; j < 4; ++j) {
      const int o = k0 + kbA + 2 * j;
      af.d[j]     = *(const unsigned int*)(Arow + o);
      af.d[4 + j] = *(const unsigned int*)(Arow + 8 + o);
    }
#pragma unroll
    for (int j = 0; j < 8; ++j) {
      const int o = k0 + kbB + 2 * j;
      b1f0.d[j] = *(const unsigned int*)(B1r + o);
      b1f1.d[j] = *(const unsigned int*)(B1r + (size_t)16 * DD + o);
      b2f0.d[j] = *(const unsigned int*)(B2r + o);
      b2f1.d[j] = *(const unsigned int*)(B2r + (size_t)16 * DD + o);
    }
    a1a = WMMA_BF16(af.v, b1f0.v, a1a);
    a1b = WMMA_BF16(af.v, b1f1.v, a1b);
    a2a = WMMA_BF16(af.v, b2f0.v, a2a);
    a2b = WMMA_BF16(af.v, b2f1.v, a2b);
  }

#pragma unroll
  for (int t = 0; t < 2; ++t) {
    const int col = nBase + t * 16 + n;
    const float bv1 = b1[e * FFF + col];
    const float bv2 = b2[e * FFF + col];
#pragma unroll
    for (int r = 0; r < 8; ++r) {
      const int rr = mBase + r + 8 * hf;
      if (rr >= count) continue;
      const float a1 = (t ? a1b[r] : a1a[r]) + bv1;
      const float a2 = (t ? a2b[r] : a2a[r]) + bv2;
      const float hv = (a1 / (1.f + __expf(-a1))) * a2;   // silu(a1)*a2
      Hbf[((size_t)e * CAP + rr) * FFF + col] = f2bf(hv);
    }
  }
}

// ---------------- expert stage 2: out += prob * (h*wp + bp); wave tile 16x64 ----------------
// grid = (DD/64, CAP/64, EE); block 128 (waves split m by 16 inside the 64-row tile; wave owns full 64 n)
__global__ __launch_bounds__(128)
void moe_out_k(const unsigned short* __restrict__ Hbf, const unsigned short* __restrict__ wpt,
               const float* __restrict__ bp, const int* __restrict__ lists,
               const float* __restrict__ probs, const int* __restrict__ cnt,
               float* __restrict__ out) {
  const int e = blockIdx.z;
  const int count = cnt[e];
  const int wave = threadIdx.x >> 5;
  const int mBase = (blockIdx.y * 4 + wave) * 16;
  if (mBase >= count) return;
  const int lane = threadIdx.x & 31;
  const int nBase = blockIdx.x * 64;
  const int m = lane & 15, hf = lane >> 4, kbA = hf * 16, kbB = hf * 16, n = lane & 15;

  const unsigned short* Arow = Hbf + ((size_t)e * CAP + mBase + m) * FFF;
  const unsigned short* Br0 = wpt + ((size_t)e * DD + nBase + n) * FFF;

  v8f acc[4] = {{}, {}, {}, {}};
  for (int k0 = 0; k0 < FFF; k0 += 32) {
    __builtin_prefetch(Arow + k0 + 512, 0, 1);
    __builtin_prefetch(Br0 + k0 + 512, 0, 1);
    BF16x16 af, bf0, bf1, bf2, bf3;
#pragma unroll
    for (int j = 0; j < 4; ++j) {
      const int o = k0 + kbA + 2 * j;
      af.d[j]     = *(const unsigned int*)(Arow + o);
      af.d[4 + j] = *(const unsigned int*)(Arow + 8 + o);
    }
#pragma unroll
    for (int j = 0; j < 8; ++j) {
      const int o = k0 + kbB + 2 * j;
      bf0.d[j] = *(const unsigned int*)(Br0 + o);
      bf1.d[j] = *(const unsigned int*)(Br0 + (size_t)16 * FFF + o);
      bf2.d[j] = *(const unsigned int*)(Br0 + (size_t)32 * FFF + o);
      bf3.d[j] = *(const unsigned int*)(Br0 + (size_t)48 * FFF + o);
    }
    acc[0] = WMMA_BF16(af.v, bf0.v, acc[0]);
    acc[1] = WMMA_BF16(af.v, bf1.v, acc[1]);
    acc[2] = WMMA_BF16(af.v, bf2.v, acc[2]);
    acc[3] = WMMA_BF16(af.v, bf3.v, acc[3]);
  }

#pragma unroll
  for (int r = 0; r < 8; ++r) {
    const int rr = mBase + r + 8 * hf;
    if (rr >= count) continue;
    const int tok = lists[e * CAP + rr];
    const float p = probs[e * CAP + rr];
#pragma unroll
    for (int t = 0; t < 4; ++t) {
      const int col = nBase + t * 16 + n;
      const float v = acc[t][r] + bp[e * DD + col];
      atomicAdd(&out[(size_t)tok * DD + col], p * v);
    }
  }
}

// ---------------- aux loss ----------------
__global__ void aux_k(const int* __restrict__ cnt, float* __restrict__ out) {
  float tot = 0.f;
  for (int e = 0; e < EE; ++e) tot += (float)cnt[e];
  float a = 0.f;
  for (int e = 0; e < EE; ++e) {
    const float d = (float)cnt[e] / tot - 1.f / (float)EE;
    a += d * d;
  }
  out[0] = a;
}

// ---------------- host launcher ----------------
extern "C" void kernel_launch(void* const* d_in, const int* in_sizes, int n_in,
                              void* d_out, int out_size, void* d_ws, size_t ws_size,
                              hipStream_t stream) {
  (void)in_sizes; (void)n_in; (void)out_size; (void)ws_size;
  const float* x      = (const float*)d_in[0];
  const float* w_qkv  = (const float*)d_in[1];
  const float* b_qkv  = (const float*)d_in[2];
  const float* w_out  = (const float*)d_in[3];
  const float* b_out  = (const float*)d_in[4];
  const float* scale1 = (const float*)d_in[5];
  const float* scale2 = (const float*)d_in[6];
  const float* w_gate = (const float*)d_in[7];
  const float* b_gate = (const float*)d_in[8];
  const float* w1     = (const float*)d_in[9];
  const float* b1     = (const float*)d_in[10];
  const float* w2     = (const float*)d_in[11];
  const float* b2     = (const float*)d_in[12];
  const float* wp     = (const float*)d_in[13];
  const float* bp     = (const float*)d_in[14];
  float* out = (float*)d_out;

  // carve workspace
  char* ws = (char*)d_ws;
  size_t off = 0;
  auto carve = [&](size_t bytes) -> char* {
    char* p = ws + off;
    off = (off + bytes + 255) & ~(size_t)255;
    return p;
  };
  unsigned short* xn1  = (unsigned short*)carve((size_t)NN * DD * 2);
  float*          qkv  = (float*)carve((size_t)NN * 3 * DD * 4);
  unsigned short* qb   = (unsigned short*)carve((size_t)NN * DD * 2);
  unsigned short* kbuf = (unsigned short*)carve((size_t)NN * DD * 2);
  unsigned short* vtb  = (unsigned short*)carve((size_t)NN * DD * 2);
  unsigned short* ob   = (unsigned short*)carve((size_t)NN * DD * 2);
  unsigned short* xn2  = (unsigned short*)carve((size_t)NN * DD * 2);
  int*            cnt   = (int*)carve(EE * sizeof(int));
  int*            lists = (int*)carve((size_t)EE * CAP * sizeof(int));
  float*          probs = (float*)carve((size_t)EE * CAP * sizeof(float));
  unsigned short* hbf   = (unsigned short*)carve((size_t)EE * CAP * FFF * 2);
  unsigned short* wqkvT = (unsigned short*)carve((size_t)DD * 3 * DD * 2);
  unsigned short* woutT = (unsigned short*)carve((size_t)DD * DD * 2);
  unsigned short* w1T   = (unsigned short*)carve((size_t)EE * DD * FFF * 2);
  unsigned short* w2T   = (unsigned short*)carve((size_t)EE * DD * FFF * 2);
  unsigned short* wpT   = (unsigned short*)carve((size_t)EE * FFF * DD * 2);

  zero_ints<<<1, 32, 0, stream>>>(cnt, EE);

  // weight prep: fp32 -> bf16 transposed [N][K]
  transpose_cvt<<<dim3(3 * DD / 32, DD / 32, 1), 256, 0, stream>>>(w_qkv, wqkvT, DD, 3 * DD);
  transpose_cvt<<<dim3(DD / 32, DD / 32, 1),     256, 0, stream>>>(w_out, woutT, DD, DD);
  transpose_cvt<<<dim3(FFF / 32, DD / 32, EE),   256, 0, stream>>>(w1, w1T, DD, FFF);
  transpose_cvt<<<dim3(FFF / 32, DD / 32, EE),   256, 0, stream>>>(w2, w2T, DD, FFF);
  transpose_cvt<<<dim3(DD / 32, FFF / 32, EE),   256, 0, stream>>>(wp, wpT, FFF, DD);

  // xn1 = rmsnorm(x, scale1) -> bf16
  rmsnorm_k<<<NN, 256, 0, stream>>>(x, scale1, xn1);

  // qkv = xn1 @ w_qkv + b_qkv
  gemm_bf16<<<dim3(3 * DD / 64, NN / 64), 128, 0, stream>>>(xn1, wqkvT, b_qkv, nullptr, qkv,
                                                            NN, 3 * DD, DD);

  // RoPE + pack q/k/vT as bf16
  rope_pack<<<dim3(TT, HH, BB), HDD, 0, stream>>>(qkv, qb, kbuf, vtb);

  // flash attention
  flash_attn<<<dim3(TT / 16, HH, BB), 32, 0, stream>>>(qb, kbuf, vtb, ob);

  // x2 = x + attn @ w_out + b_out   (written straight into d_out)
  gemm_bf16<<<dim3(DD / 64, NN / 64), 128, 0, stream>>>(ob, woutT, b_out, x, out, NN, DD, DD);

  // xn2 = rmsnorm(x2, scale2) -> bf16
  rmsnorm_k<<<NN, 256, 0, stream>>>(out, scale2, xn2);

  // gating / expert lists
  gate_topk<<<NN, 32, 0, stream>>>(xn2, w_gate, b_gate, cnt, lists, probs);

  // expert FF stage 1 (w1/w2 fused, SiLU-gated)
  moe_ff1<<<dim3(FFF / 128, CAP / 16, EE), 128, 0, stream>>>(xn2, w1T, b1, w2T, b2,
                                                             lists, cnt, hbf);

  // expert FF stage 2 (wp) + weighted scatter-add into output
  moe_out_k<<<dim3(DD / 64, CAP / 64, EE), 128, 0, stream>>>(hbf, wpT, bp, lists, probs,
                                                             cnt, out);

  // aux loss scalar
  aux_k<<<1, 1, 0, stream>>>(cnt, out + (size_t)NN * DD);
}
